// EquivariantSUG_27891517620928
// MI455X (gfx1250) — matrix operand order repaired and checked by reading, not compiled, not run on hardware
//
#include <hip/hip_runtime.h>

#define NN 20000
#define EEDG 320000
#define BB 16
#define IND 28
#define HD 128
#define HH 64
#define OUTD 64
#define NBD 32
#define LAY 4
#define FEATD 416
#define PI_F 3.14159265358979f

// packed-weight offsets (in halfs) within one layer's block
#define PM1 0u
#define PM2 53248u
#define PM3 69632u
#define PA1 86016u
#define PC1 112640u
#define PC2 165888u
#define PW_LAYER 182272u

// edge-MLP tiling: 2 waves/block, 2 independent 16-edge tiles per wave
#define WPB 2
#define TPW 2
#define MI_TS 6656            // 16 x 416 halfs per mi tile
#define T_TS 2048             // 16 x 128 halfs per temp tile
#define WV_HALFS (TPW * MI_TS + 2 * TPW * T_TS)   // 21504 halfs per wave
#define SMEM_BYTES (WPB * WV_HALFS * 2)           // 86016 B (dynamic LDS)

typedef __attribute__((ext_vector_type(16))) __bf16 v16bf;
typedef __attribute__((ext_vector_type(8)))  float  v8f;

__device__ __forceinline__ unsigned short bfbits(float f) {
  unsigned u = __builtin_bit_cast(unsigned, f);
  u += 0x7fffu + ((u >> 16) & 1u);
  return (unsigned short)(u >> 16);
}
__device__ __forceinline__ float bf2f(unsigned short h) {
  unsigned u = ((unsigned)h) << 16;
  return __builtin_bit_cast(float, u);
}
__device__ __forceinline__ float siluf(float v) { return v / (1.f + __expf(-v)); }

// ---------------------------------------------------------------------------
// Multi-tile 16xK @ KxN bf16 WMMA GEMM for one wave.
// T independent 16-row tiles share every B tile (in-wave B reuse);
// A fragments are reused across an nt-group of 4.
// All LDS addresses are one base register per tile + compile-time immediates:
//   A base = src + m*LD_SRC + 8*hf  (A layout: VGPR j<4: K=8*hf+2j;
//                                    j>=4: K=16+8*hf+2(j-4))
//   D base = dst + n + 8*hf*LD_DST  (C layout: VGPR r -> M=r+8*hf, N=n)
// pw: packed B tiles [KT][NT][32 lanes][16 halfs]
// ---------------------------------------------------------------------------
template <int T, int KT, int NT, int LD_SRC, int LD_DST, bool SILU>
__device__ __forceinline__ void wmma_gemmT(const unsigned short* src, int src_ts,
                                           const unsigned short* pw,
                                           const float* bias, unsigned short* dst,
                                           long dst_ts, int lane) {
  const int n = lane & 15, hf = lane >> 4;
  const unsigned short* ab[T];
  unsigned short* db[T];
#pragma unroll
  for (int t = 0; t < T; ++t) {
    ab[t] = src + (size_t)t * src_ts + n * LD_SRC + 8 * hf;
    db[t] = dst + (long)t * dst_ts + n + (size_t)(8 * hf) * LD_DST;
  }
  const unsigned short* pb = pw + (size_t)lane * 16;
  constexpr int NTH = 4;
  constexpr int NP = NT / NTH;
#pragma unroll
  for (int p = 0; p < NP; ++p) {
    v8f acc[T][NTH];
#pragma unroll
    for (int q = 0; q < NTH; ++q) {
      float bv = bias[(p * NTH + q) * 16 + n];
#pragma unroll
      for (int t = 0; t < T; ++t)
#pragma unroll
        for (int r = 0; r < 8; ++r) acc[t][q][r] = bv;
    }
#pragma unroll
    for (int kt = 0; kt < KT; ++kt) {
      v16bf a[T];
#pragma unroll
      for (int t = 0; t < T; ++t) {
#pragma unroll
        for (int j = 0; j < 8; ++j) {
          const int o = kt * 32 + ((j & 4) ? 16 : 0) + 2 * (j & 3);  // const imm
          unsigned v = *(const unsigned*)(ab[t] + o);
          a[t][2 * j]     = __builtin_bit_cast(__bf16, (unsigned short)(v & 0xffffu));
          a[t][2 * j + 1] = __builtin_bit_cast(__bf16, (unsigned short)(v >> 16));
        }
      }
#pragma unroll
      for (int q = 0; q < NTH; ++q) {
        const int nt = p * NTH + q;
        v16bf b = *(const v16bf*)(pb + (size_t)(kt * NT + nt) * 32 * 16);
#pragma unroll
        for (int t = 0; t < T; ++t)
          acc[t][q] = __builtin_amdgcn_wmma_f32_16x16x32_bf16(
              false, a[t], false, b, (short)0, acc[t][q], false, false);
      }
    }
#pragma unroll
    for (int q = 0; q < NTH; ++q) {
      const int nt = p * NTH + q;
#pragma unroll
      for (int t = 0; t < T; ++t)
#pragma unroll
        for (int r = 0; r < 8; ++r) {
          float v = acc[t][q][r];
          if (SILU) v = siluf(v);
          db[t][r * LD_DST + nt * 16] = bfbits(v);   // const imm offsets
        }
    }
  }
}

// ---------------------------------------------------------------------------
// Weight repack: fp32 (Kdim x Ndim) row-major -> bf16 B-operand tiles
// B layout: lane -> N = nt*16 + (lane&15); element e -> K = kt*32 + 8*(lane>>4) + (e&7) + 16*(e>>3)
// ---------------------------------------------------------------------------
__global__ void k_pack(const float* __restrict__ W, unsigned short* __restrict__ out,
                       int Kdim, int Ndim) {
  int ntt = Ndim / 16;
  int kt = blockIdx.x / ntt, nt = blockIdx.x % ntt;
  int lane = threadIdx.x;
  int Nc = nt * 16 + (lane & 15);
  unsigned short* dst = out + ((size_t)(kt * ntt + nt) * 32 + lane) * 16;
#pragma unroll
  for (int e = 0; e < 16; ++e) {
    int K = kt * 32 + 8 * (lane >> 4) + (e & 7) + ((e & 8) ? 16 : 0);
    dst[e] = bfbits(W[(size_t)K * Ndim + Nc]);
  }
}

__global__ void k_zero(float* p, long n) {
  long i = (long)blockIdx.x * 256 + threadIdx.x;
  if (i < n) p[i] = 0.f;
}

// h = x @ in_w + in_b ; also bf16 copy, and cur = pos
__global__ __launch_bounds__(128) void k_embed(const float* __restrict__ x,
                                               const float* __restrict__ pos,
                                               const float* __restrict__ w,
                                               const float* __restrict__ b,
                                               float* __restrict__ h,
                                               unsigned short* __restrict__ h_bf,
                                               float* __restrict__ cur) {
  int n = blockIdx.x, j = threadIdx.x;
  __shared__ float xs[IND];
  if (j < IND) xs[j] = x[(size_t)n * IND + j];
  __syncthreads();
  float s = b[j];
#pragma unroll
  for (int k = 0; k < IND; ++k) s += xs[k] * w[k * HD + j];
  h[(size_t)n * HD + j] = s;
  h_bf[(size_t)n * HD + j] = bfbits(s);
  if (j < 3) cur[n * 3 + j] = pos[n * 3 + j];
}

// edge features ef = [rbf(32)*cutoff, sh(128)]  (bf16, fixed across layers)
__global__ void k_edge_geom(const float* __restrict__ pos, const int* __restrict__ row,
                            const int* __restrict__ col, const float* __restrict__ rbf_c,
                            const float* __restrict__ rbf_w, const float* __restrict__ sh_w,
                            const float* __restrict__ sh_b, unsigned short* __restrict__ ef) {
  int e = blockIdx.x * 256 + threadIdx.x;
  if (e >= EEDG) return;
  int r = row[e], c = col[e];
  float ex = pos[c * 3 + 0] - pos[r * 3 + 0];
  float ey = pos[c * 3 + 1] - pos[r * 3 + 1];
  float ez = pos[c * 3 + 2] - pos[r * 3 + 2];
  float d = sqrtf(ex * ex + ey * ey + ez * ez);
  float cf = (d < 10.f) ? 0.5f * (__cosf(PI_F * d * 0.1f) + 1.f) : 0.f;
  unsigned short* dst = ef + (size_t)e * 160;
#pragma unroll
  for (int b = 0; b < NBD; ++b) {
    float dc = d - rbf_c[b];
    float wv = rbf_w[b];
    float v = __expf(-dc * dc / (2.f * wv * wv)) * cf;
    dst[b] = bfbits(v);
  }
  float dinv = 1.f / fmaxf(d, 1e-12f);
  float dx = ex * dinv, dy = ey * dinv, dz = ez * dinv;
  float sv[8] = {dx, dy, dz, dx * dy, dy * dz, dz * dz - 0.5f * (dx * dx + dy * dy),
                 dx * dz, dx * dx - dy * dy};
  for (int j = 0; j < HD; ++j) {
    float s = sh_b[j];
#pragma unroll
    for (int k = 0; k < 8; ++k) s += sv[k] * sh_w[k * HD + j];
    dst[NBD + j] = bfbits(s);
  }
}

// ---------------------------------------------------------------------------
// THE hot kernel: per-edge MLPs via bf16 WMMA.
// One wave = 2 independent 16-edge tiles (in-wave B reuse x2, A reuse x4).
// Dynamic LDS: per wave [mi 2x(16x416) | tA 2x(16x128) | tB 2x(16x128)] bf16.
// mi gather uses CDNA5 async global->LDS copies (ASYNCcnt), no VGPR staging.
// ---------------------------------------------------------------------------
__global__ __launch_bounds__(64) void k_edge_mlp(
    const unsigned short* __restrict__ h_bf, const unsigned short* __restrict__ ef,
    const int* __restrict__ row, const int* __restrict__ col,
    const unsigned short* __restrict__ pw, const float* __restrict__ b_msg1,
    const float* __restrict__ b_msg2, const float* __restrict__ b_msg3,
    const float* __restrict__ b_att1, const float* __restrict__ att_w2,
    const float* __restrict__ att_b2, const float* __restrict__ b_crd1,
    const float* __restrict__ b_crd2, const float* __restrict__ crd_w3,
    const float* __restrict__ crd_b3, unsigned short* __restrict__ sm_out,
    float* __restrict__ al_out, float* __restrict__ cw_out) {
  extern __shared__ unsigned short smem[];
  const int w = threadIdx.x >> 5, lane = threadIdx.x & 31;
  const int e0 = (blockIdx.x * WPB + w) * (TPW * 16);
  if (e0 >= EEDG) return;
  unsigned short* MI = smem + (size_t)w * WV_HALFS;
  unsigned short* TA = MI + TPW * MI_TS;
  unsigned short* TB = TA + TPW * T_TS;

  // gather mi = [h[row] | h[col] | ef]: 32 lanes <-> 32 edge rows (2 tiles).
  // Async global->LDS: one LDS-addr reg + one 64-bit global-addr reg per
  // stream; inst offset applies to both sides. Single s_wait_asynccnt after.
  {
    const int t = lane >> 4, m = lane & 15;
    const int e = e0 + lane;
    const int r = row[e], c = col[e];
    // dynamic LDS starts at offset 0 (no static LDS in this kernel)
    unsigned lds0 =
        (unsigned)(((size_t)w * WV_HALFS + (size_t)t * MI_TS + (size_t)m * FEATD) * 2);
    unsigned lds1 = lds0 + 256u;  // h[col] region
    unsigned lds2 = lds0 + 512u;  // ef region
    unsigned long long ga_r = (unsigned long long)(h_bf + (size_t)r * HD);
    unsigned long long ga_c = (unsigned long long)(h_bf + (size_t)c * HD);
    unsigned long long ga_e = (unsigned long long)(ef + (size_t)e * 160);
#pragma unroll
    for (int i = 0; i < 16; ++i)
      asm volatile("global_load_async_to_lds_b128 %0, %1, off offset:%2"
                   :: "v"(lds0), "v"(ga_r), "i"(i * 16));
#pragma unroll
    for (int i = 0; i < 16; ++i)
      asm volatile("global_load_async_to_lds_b128 %0, %1, off offset:%2"
                   :: "v"(lds1), "v"(ga_c), "i"(i * 16));
#pragma unroll
    for (int i = 0; i < 20; ++i)
      asm volatile("global_load_async_to_lds_b128 %0, %1, off offset:%2"
                   :: "v"(lds2), "v"(ga_e), "i"(i * 16));
    asm volatile("s_wait_asynccnt 0" ::: "memory");
  }
  // message MLP: 416->128 (silu) ->128 (silu) ->128
  wmma_gemmT<TPW, 13, 8, FEATD, HD, true >(MI, MI_TS, pw + PM1, b_msg1, TA, T_TS, lane);
  wmma_gemmT<TPW, 4,  8, HD,    HD, true >(TA, T_TS, pw + PM2, b_msg2, TB, T_TS, lane);
  wmma_gemmT<TPW, 4,  8, HD,    HD, false>(TB, T_TS, pw + PM3, b_msg3,
                                           sm_out + (size_t)e0 * HD, 16L * HD, lane);
  // attention: 416->64 (silu) -> 1
  wmma_gemmT<TPW, 13, 4, FEATD, HH, true>(MI, MI_TS, pw + PA1, b_att1, TA, T_TS, lane);
  if (lane < 16) {
#pragma unroll
    for (int t = 0; t < TPW; ++t) {
      float s = att_b2[0];
#pragma unroll
      for (int k = 0; k < HH; ++k) s += bf2f(TA[t * T_TS + lane * HH + k]) * att_w2[k];
      al_out[e0 + t * 16 + lane] = s;
    }
  }
  // coord MLP: 416->128 (silu) ->128 (silu) -> 1
  wmma_gemmT<TPW, 13, 8, FEATD, HD, true>(MI, MI_TS, pw + PC1, b_crd1, TA, T_TS, lane);
  wmma_gemmT<TPW, 4,  8, HD,    HD, true>(TA, T_TS, pw + PC2, b_crd2, TB, T_TS, lane);
  if (lane < 16) {
#pragma unroll
    for (int t = 0; t < TPW; ++t) {
      float s = crd_b3[0];
#pragma unroll
      for (int k = 0; k < HD; ++k) s += bf2f(TB[t * T_TS + lane * HD + k]) * crd_w3[k];
      cw_out[e0 + t * 16 + lane] = s;
    }
  }
}

// scatter-softmax pass 1: mx[c] = max(max_e al, 0)  (int-punned atomicMax, init 0)
__global__ void k_att_max(const float* __restrict__ al, const int* __restrict__ col,
                          float* __restrict__ mx) {
  int e = blockIdx.x * 256 + threadIdx.x;
  if (e >= EEDG) return;
  float v = al[e];
  if (v > 0.f) atomicMax((int*)&mx[col[e]], __float_as_int(v));
}

// pass 2: ex = exp(al - mx[c]); den[c] += ex (al buffer overwritten with ex)
__global__ void k_att_sum(float* __restrict__ alx, const float* __restrict__ mx,
                          float* __restrict__ den, const int* __restrict__ col) {
  int e = blockIdx.x * 256 + threadIdx.x;
  if (e >= EEDG) return;
  int c = col[e];
  float ex = __expf(alx[e] - mx[c]);
  alx[e] = ex;
  atomicAdd(&den[c], ex);
}

// pass 3: agg[c] += sm * softmax ; dpos[c] += cw * dir(cur).  1 wave per edge.
__global__ __launch_bounds__(256) void k_scatter(
    const unsigned short* __restrict__ sm, const float* __restrict__ exv,
    const float* __restrict__ den, const float* __restrict__ cwv,
    const int* __restrict__ row, const int* __restrict__ col,
    const float* __restrict__ cur, float* __restrict__ agg, float* __restrict__ dpos) {
  int e = blockIdx.x * 8 + (threadIdx.x >> 5);
  if (e >= EEDG) return;
  int lane = threadIdx.x & 31;
  int c = col[e];
  float wgt = exv[e] / (den[c] + 1e-8f);
  const unsigned short* sme = sm + (size_t)e * HD;
#pragma unroll
  for (int i = 0; i < 4; ++i) {
    int k = lane * 4 + i;
    atomicAdd(&agg[(size_t)c * HD + k], bf2f(sme[k]) * wgt);
  }
  if (lane == 0) {
    int r = row[e];
    float ex = cur[c * 3 + 0] - cur[r * 3 + 0];
    float ey = cur[c * 3 + 1] - cur[r * 3 + 1];
    float ez = cur[c * 3 + 2] - cur[r * 3 + 2];
    float dl = sqrtf(ex * ex + ey * ey + ez * ez);
    float inv = 1.f / (dl + 1e-8f);
    float cv = cwv[e];
    atomicAdd(&dpos[c * 3 + 0], cv * ex * inv);
    atomicAdd(&dpos[c * 3 + 1], cv * ey * inv);
    atomicAdd(&dpos[c * 3 + 2], cv * ez * inv);
  }
}

// node update MLP + residual layernorm + coordinate integrate
__global__ __launch_bounds__(128) void k_node_update(
    float* __restrict__ h, unsigned short* __restrict__ h_bf,
    const float* __restrict__ agg, const float* __restrict__ w1,
    const float* __restrict__ b1, const float* __restrict__ w2,
    const float* __restrict__ b2, const float* __restrict__ g,
    const float* __restrict__ bb, float* __restrict__ cur,
    const float* __restrict__ dpos) {
  int n = blockIdx.x, j = threadIdx.x;
  __shared__ float inb[2 * HD];
  __shared__ float u1[HD];
  __shared__ float red[HD];
  float hj = h[(size_t)n * HD + j];
  inb[j] = hj;
  inb[HD + j] = agg[(size_t)n * HD + j];
  __syncthreads();
  float s = b1[j];
  for (int k = 0; k < 2 * HD; ++k) s += inb[k] * w1[k * HD + j];
  u1[j] = siluf(s);
  __syncthreads();
  float s2 = b2[j];
  for (int k = 0; k < HD; ++k) s2 += u1[k] * w2[k * HD + j];
  float hv = hj + s2;
  red[j] = hv;
  __syncthreads();
  for (int st = 64; st > 0; st >>= 1) { if (j < st) red[j] += red[j + st]; __syncthreads(); }
  float mean = red[0] * (1.f / HD);
  __syncthreads();
  float dv = hv - mean;
  red[j] = dv * dv;
  __syncthreads();
  for (int st = 64; st > 0; st >>= 1) { if (j < st) red[j] += red[j + st]; __syncthreads(); }
  float var = red[0] * (1.f / HD);
  float o = dv * rsqrtf(var + 1e-5f) * g[j] + bb[j];
  h[(size_t)n * HD + j] = o;
  h_bf[(size_t)n * HD + j] = bfbits(o);
  if (j < 3) cur[n * 3 + j] += dpos[n * 3 + j];
}

// pooling attention scores (global + lysine heads)
__global__ __launch_bounds__(64) void k_scores(
    const float* __restrict__ h, const float* __restrict__ ga_w1,
    const float* __restrict__ ga_b1, const float* __restrict__ ga_w2,
    const float* __restrict__ ga_b2, const float* __restrict__ la_w1,
    const float* __restrict__ la_b1, const float* __restrict__ la_w2,
    const float* __restrict__ la_b2, float* __restrict__ sg, float* __restrict__ sl) {
  int n = blockIdx.x, t = threadIdx.x;
  __shared__ float hs[HD];
  __shared__ float pr[HH];
  hs[t] = h[(size_t)n * HD + t];
  hs[HH + t] = h[(size_t)n * HD + HH + t];
  __syncthreads();
  float s = ga_b1[t];
  for (int k = 0; k < HD; ++k) s += hs[k] * ga_w1[k * HH + t];
  pr[t] = tanhf(s) * ga_w2[t];
  __syncthreads();
  for (int st = 32; st > 0; st >>= 1) { if (t < st) pr[t] += pr[t + st]; __syncthreads(); }
  if (t == 0) sg[n] = pr[0] + ga_b2[0];
  __syncthreads();
  s = la_b1[t];
  for (int k = 0; k < HD; ++k) s += hs[k] * la_w1[k * HH + t];
  pr[t] = tanhf(s) * la_w2[t];
  __syncthreads();
  for (int st = 32; st > 0; st >>= 1) { if (t < st) pr[t] += pr[t + st]; __syncthreads(); }
  if (t == 0) sl[n] = pr[0] + la_b2[0];
}

__global__ void k_bmax(const float* __restrict__ sg, const float* __restrict__ sl,
                       const unsigned char* __restrict__ mask, const int* __restrict__ batch,
                       float* __restrict__ mxg, float* __restrict__ mxl) {
  int n = blockIdx.x * 256 + threadIdx.x;
  if (n >= NN) return;
  int b = batch[n];
  float v = sg[n];
  if (v > 0.f) atomicMax((int*)&mxg[b], __float_as_int(v));
  if (mask[n]) {
    float u = sl[n];
    if (u > 0.f) atomicMax((int*)&mxl[b], __float_as_int(u));
  }
}

__global__ void k_bsum(float* __restrict__ sg, float* __restrict__ sl,
                       const float* __restrict__ mxg, const float* __restrict__ mxl,
                       float* __restrict__ deng, float* __restrict__ denl,
                       float* __restrict__ cnt, const unsigned char* __restrict__ mask,
                       const int* __restrict__ batch) {
  int n = blockIdx.x * 256 + threadIdx.x;
  if (n >= NN) return;
  int b = batch[n];
  float eg = __expf(sg[n] - mxg[b]);
  sg[n] = eg;
  atomicAdd(&deng[b], eg);
  if (mask[n]) {
    float el = __expf(sl[n] - mxl[b]);
    sl[n] = el;
    atomicAdd(&denl[b], el);
  } else {
    sl[n] = 0.f;
  }
  atomicAdd(&cnt[b], 1.f);
}

__global__ __launch_bounds__(128) void k_pool(
    const float* __restrict__ h, const float* __restrict__ exg,
    const float* __restrict__ exl, const float* __restrict__ deng,
    const float* __restrict__ denl, const int* __restrict__ batch,
    float* __restrict__ gp, float* __restrict__ lp) {
  int n = blockIdx.x, j = threadIdx.x;
  int b = batch[n];
  float hv = h[(size_t)n * HD + j];
  float aw = exg[n] / (deng[b] + 1e-8f);
  float lw = exl[n] / (denl[b] + 1e-8f);
  atomicAdd(&gp[b * HD + j], hv * aw);
  atomicAdd(&lp[b * HD + j], hv * lw);
}

__global__ __launch_bounds__(128) void k_head(
    const float* __restrict__ gp, const float* __restrict__ lp,
    const float* __restrict__ cnt, const float* __restrict__ pool_w,
    const float* __restrict__ pool_b, const float* __restrict__ out_w,
    const float* __restrict__ out_b, float* __restrict__ out) {
  int b = blockIdx.x, j = threadIdx.x;
  __shared__ float pl[2 * HD];
  __shared__ float ph[HD];
  float inv = rsqrtf(fmaxf(cnt[b], 1.f));
  pl[j] = gp[b * HD + j] * inv;
  pl[HD + j] = lp[b * HD + j] * inv;
  __syncthreads();
  float s = pool_b[j];
  for (int k = 0; k < 2 * HD; ++k) s += pl[k] * pool_w[k * HD + j];
  ph[j] = s;
  __syncthreads();
  if (j < OUTD) {
    float o = out_b[j];
    for (int k = 0; k < HD; ++k) o += ph[k] * out_w[k * OUTD + j];
    out[b * OUTD + j] = o;
  }
}

// ---------------------------------------------------------------------------
extern "C" void kernel_launch(void* const* d_in, const int* in_sizes, int n_in,
                              void* d_out, int out_size, void* d_ws, size_t ws_size,
                              hipStream_t stream) {
  (void)in_sizes; (void)n_in; (void)out_size; (void)ws_size;
  // inputs (setup_inputs dict order, params flattened in insertion order)
  const float* x      = (const float*)d_in[0];
  const float* pos    = (const float*)d_in[1];
  const int*   ei     = (const int*)d_in[2];
  const int*   row    = ei;
  const int*   col    = ei + EEDG;
  const int*   batch  = (const int*)d_in[3];
  const unsigned char* mask = (const unsigned char*)d_in[4];
  const float* in_w   = (const float*)d_in[5];
  const float* in_b   = (const float*)d_in[6];
  const float* rbf_c  = (const float*)d_in[7];
  const float* rbf_w  = (const float*)d_in[8];
  const float* sh_w   = (const float*)d_in[9];
  const float* sh_b   = (const float*)d_in[10];
  const float* msg_w1 = (const float*)d_in[11];
  const float* msg_b1 = (const float*)d_in[12];
  const float* msg_w2 = (const float*)d_in[13];
  const float* msg_b2 = (const float*)d_in[14];
  const float* msg_w3 = (const float*)d_in[15];
  const float* msg_b3 = (const float*)d_in[16];
  const float* att_w1 = (const float*)d_in[17];
  const float* att_b1 = (const float*)d_in[18];
  const float* att_w2 = (const float*)d_in[19];
  const float* att_b2 = (const float*)d_in[20];
  const float* crd_w1 = (const float*)d_in[21];
  const float* crd_b1 = (const float*)d_in[22];
  const float* crd_w2 = (const float*)d_in[23];
  const float* crd_b2 = (const float*)d_in[24];
  const float* crd_w3 = (const float*)d_in[25];
  const float* crd_b3 = (const float*)d_in[26];
  const float* upd_w1 = (const float*)d_in[27];
  const float* upd_b1 = (const float*)d_in[28];
  const float* upd_w2 = (const float*)d_in[29];
  const float* upd_b2 = (const float*)d_in[30];
  const float* ln_g   = (const float*)d_in[31];
  const float* ln_b   = (const float*)d_in[32];
  const float* ga_w1  = (const float*)d_in[33];
  const float* ga_b1  = (const float*)d_in[34];
  const float* ga_w2  = (const float*)d_in[35];
  const float* ga_b2  = (const float*)d_in[36];
  const float* la_w1  = (const float*)d_in[37];
  const float* la_b1  = (const float*)d_in[38];
  const float* la_w2  = (const float*)d_in[39];
  const float* la_b2  = (const float*)d_in[40];
  const float* pool_w = (const float*)d_in[41];
  const float* pool_b = (const float*)d_in[42];
  const float* out_w  = (const float*)d_in[43];
  const float* out_b  = (const float*)d_in[44];
  float* out = (float*)d_out;

  // workspace partition (256B aligned)
  char* ws = (char*)d_ws;
  size_t off = 0;
  auto alloc = [&](size_t bytes) {
    size_t o = off;
    off = (off + bytes + 255) & ~(size_t)255;
    return o;
  };
  float*          h     = (float*)(ws + alloc((size_t)NN * HD * 4));
  unsigned short* h_bf  = (unsigned short*)(ws + alloc((size_t)NN * HD * 2));
  float*          cur   = (float*)(ws + alloc((size_t)NN * 3 * 4));
  unsigned short* efb   = (unsigned short*)(ws + alloc((size_t)EEDG * 160 * 2));
  unsigned short* smb   = (unsigned short*)(ws + alloc((size_t)EEDG * HD * 2));
  float*          alx   = (float*)(ws + alloc((size_t)EEDG * 4));
  float*          cwb   = (float*)(ws + alloc((size_t)EEDG * 4));
  float*          scat  = (float*)(ws + alloc((size_t)NN * (2 + HD + 3) * 4));
  float*          mx    = scat;
  float*          den   = scat + NN;
  float*          agg   = scat + 2 * NN;
  float*          dpos  = scat + (size_t)(2 + HD) * NN;
  unsigned short* pw    = (unsigned short*)(ws + alloc((size_t)LAY * PW_LAYER * 2));
  float*          sg    = (float*)(ws + alloc((size_t)NN * 4));
  float*          sl    = (float*)(ws + alloc((size_t)NN * 4));
  float*          bpool = (float*)(ws + alloc((size_t)(BB * (5 + 2 * HD)) * 4));
  float*          mxg   = bpool;
  float*          deng  = bpool + BB;
  float*          mxl   = bpool + 2 * BB;
  float*          denl  = bpool + 3 * BB;
  float*          cnt   = bpool + 4 * BB;
  float*          gp    = bpool + 5 * BB;
  float*          lp    = bpool + 5 * BB + BB * HD;

  // pack all edge-MLP weights into WMMA B-operand layout (idempotent)
  for (int l = 0; l < LAY; ++l) {
    unsigned short* base = pw + (size_t)l * PW_LAYER;
    k_pack<<<13 * 8, 32, 0, stream>>>(msg_w1 + (size_t)l * FEATD * HD, base + PM1, FEATD, HD);
    k_pack<<<4 * 8, 32, 0, stream>>>(msg_w2 + (size_t)l * HD * HD, base + PM2, HD, HD);
    k_pack<<<4 * 8, 32, 0, stream>>>(msg_w3 + (size_t)l * HD * HD, base + PM3, HD, HD);
    k_pack<<<13 * 4, 32, 0, stream>>>(att_w1 + (size_t)l * FEATD * HH, base + PA1, FEATD, HH);
    k_pack<<<13 * 8, 32, 0, stream>>>(crd_w1 + (size_t)l * FEATD * HD, base + PC1, FEATD, HD);
    k_pack<<<4 * 8, 32, 0, stream>>>(crd_w2 + (size_t)l * HD * HD, base + PC2, HD, HD);
  }

  k_embed<<<NN, 128, 0, stream>>>(x, pos, in_w, in_b, h, h_bf, cur);
  k_edge_geom<<<(EEDG + 255) / 256, 256, 0, stream>>>(pos, row, col, rbf_c, rbf_w,
                                                      sh_w, sh_b, efb);

  const int mlp_blocks = EEDG / (WPB * TPW * 16);  // 5000
  for (int l = 0; l < LAY; ++l) {
    long zc = (long)NN * (2 + HD + 3);
    k_zero<<<(unsigned)((zc + 255) / 256), 256, 0, stream>>>(scat, zc);
    k_edge_mlp<<<mlp_blocks, WPB * 32, SMEM_BYTES, stream>>>(
        h_bf, efb, row, col, pw + (size_t)l * PW_LAYER, msg_b1 + l * HD,
        msg_b2 + l * HD, msg_b3 + l * HD, att_b1 + l * HH, att_w2 + l * HH,
        att_b2 + l, crd_b1 + l * HD, crd_b2 + l * HD, crd_w3 + l * HD, crd_b3 + l,
        smb, alx, cwb);
    k_att_max<<<(EEDG + 255) / 256, 256, 0, stream>>>(alx, col, mx);
    k_att_sum<<<(EEDG + 255) / 256, 256, 0, stream>>>(alx, mx, den, col);
    k_scatter<<<(EEDG + 7) / 8, 256, 0, stream>>>(smb, alx, den, cwb, row, col, cur,
                                                  agg, dpos);
    k_node_update<<<NN, 128, 0, stream>>>(h, h_bf, agg, upd_w1 + (size_t)l * 2 * HD * HD,
                                          upd_b1 + l * HD, upd_w2 + (size_t)l * HD * HD,
                                          upd_b2 + l * HD, ln_g + l * HD, ln_b + l * HD,
                                          cur, dpos);
  }

  k_scores<<<NN, 64, 0, stream>>>(h, ga_w1, ga_b1, ga_w2, ga_b2, la_w1, la_b1, la_w2,
                                  la_b2, sg, sl);
  long bz = (long)BB * (5 + 2 * HD);
  k_zero<<<(unsigned)((bz + 255) / 256), 256, 0, stream>>>(bpool, bz);
  k_bmax<<<(NN + 255) / 256, 256, 0, stream>>>(sg, sl, mask, batch, mxg, mxl);
  k_bsum<<<(NN + 255) / 256, 256, 0, stream>>>(sg, sl, mxg, mxl, deng, denl, cnt, mask,
                                               batch);
  k_pool<<<NN, 128, 0, stream>>>(h, sg, sl, deng, denl, batch, gp, lp);
  k_head<<<BB, 128, 0, stream>>>(gp, lp, cnt, pool_w, pool_b, out_w, out_b, out);
}
